// MultiScaleAttentionNew_54279796686852
// MI455X (gfx1250) — matrix operand order repaired and checked by reading, compiled, not run on hardware
//
#include <hip/hip_runtime.h>
#include <hip/hip_bf16.h>
#include <math.h>

// ---------------------------------------------------------------------------
// MViT-style multi-scale attention for MI455X (gfx1250, wave32, WMMA).
// All GEMM-shaped math on v_wmma_f32_16x16x32_bf16; every WMMA operand is a
// contiguous unguarded 16B vector load; GEMM waves compute 16x64 tiles with
// all four B fragments in flight so loads overlap the matrix pipe.
// ---------------------------------------------------------------------------

typedef __bf16 bf16_t;
typedef __attribute__((ext_vector_type(16))) __bf16 v16bf;
typedef __attribute__((ext_vector_type(8)))  __bf16 v8bf;
typedef __attribute__((ext_vector_type(8)))  float  v8f;

#define BATCH   8
#define CDIM    768
#define HEADS   12
#define HD      64
#define NSPAT   1568            // 8*14*14
#define NTOK    1569            // cls + spatial
#define TOTM    (BATCH * NTOK)  // 12552 valid GEMM rows
#define MPAD    12576           // TOTM padded to multiple of 32 (393 macro tiles)
#define NTOKP   1584            // NTOK padded to multiple of 16 (99 tiles)
#define LKV     393             // cls + 8*7*7
#define LKP     416             // LKV padded to 13*32
#define SCALE   0.125f          // 64^-0.5
#define XOUT_ELEMS ((size_t)BATCH * CDIM * NSPAT)

// A-fragment (bf16 16x32, MxK): lanes 0-15 hold K {0..7,16..23} of row=lane,
// lanes 16-31 hold K {8..15,24..31} of row=lane-16. Two contiguous 8-elt runs.
__device__ __forceinline__ v8f wmma_bf16(v16bf a, v16bf b, v8f c) {
    return __builtin_amdgcn_wmma_f32_16x16x32_bf16(
        false, a, false, b, (short)0, c, false, false);
}

__device__ __forceinline__ v16bf make_frag(v8bf lo, v8bf hi) {
    v16bf r;
#pragma unroll
    for (int e = 0; e < 8; ++e) { r[e] = lo[e]; r[e + 8] = hi[e]; }
    return r;
}

// ------------------------- small helpers -----------------------------------
__global__ __launch_bounds__(256) void cvt_bf16_kernel(
    const float* __restrict__ src, bf16_t* __restrict__ dst, int n) {
    int i = blockIdx.x * 256 + threadIdx.x;
    if (i < n) dst[i] = (bf16_t)src[i];
}

__global__ __launch_bounds__(256) void zero_bf16_kernel(bf16_t* p, int n) {
    int i = blockIdx.x * 256 + threadIdx.x;
    if (i < n) p[i] = (bf16_t)0.f;
}

// ---------------- pack x -> dense bf16 A (MPAD x CDIM) ---------------------
// x is (B, C, 1568): transpose (C, sp) -> (tok, C) via LDS tile.
__global__ __launch_bounds__(256) void pack_x_kernel(
    const float* __restrict__ x, bf16_t* __restrict__ Ab) {
    __shared__ float tile[32][33];
    const int sp0 = blockIdx.x * 32;   // 49 tiles
    const int k0  = blockIdx.y * 32;   // 24 tiles
    const int b   = blockIdx.z;
    const int tx = threadIdx.x, ty = threadIdx.y;   // (32, 8)
#pragma unroll
    for (int j = 0; j < 4; ++j) {
        int kr = ty + j * 8;
        tile[kr][tx] = x[((size_t)b * CDIM + k0 + kr) * NSPAT + sp0 + tx];
    }
    __syncthreads();
#pragma unroll
    for (int j = 0; j < 4; ++j) {
        int sr = ty + j * 8;
        Ab[((size_t)b * NTOK + 1 + sp0 + sr) * CDIM + k0 + tx] = (bf16_t)tile[tx][sr];
    }
}

// cls rows -> A row b*NTOK; zero the M-pad rows [TOTM, MPAD)
__global__ __launch_bounds__(256) void pack_cls_pad_kernel(
    const float* __restrict__ cls, bf16_t* __restrict__ Ab) {
    int i = blockIdx.x * 256 + threadIdx.x;
    if (i < BATCH * CDIM) {
        int b = i / CDIM, c = i % CDIM;
        Ab[(size_t)b * NTOK * CDIM + c] = (bf16_t)cls[i];
    }
    if (i < (MPAD - TOTM) * CDIM)
        Ab[(size_t)TOTM * CDIM + i] = (bf16_t)0.f;
}

// ------------------------- WMMA GEMM ---------------------------------------
// Y[m,n] = sum_k A[m,k] * W[n,k] + bias[n];  A dense bf16 (MPAD x CDIM).
// mode 0: Y -> (B,NTOK,C) f32.  mode 1: scatter to d_out (xo + cls tail).
// Block = 256 = 8 waves laid out 2(M) x 4(N); each wave owns a 16x64 tile.
// All four B fragments are loaded before the four WMMAs so the compiler can
// use partial s_wait_loadcnt and overlap VMEM with the matrix pipe.
__global__ __launch_bounds__(256) void wmma_gemm_kernel(
    const bf16_t* __restrict__ Ab,
    const bf16_t* __restrict__ Wb,
    const float*  __restrict__ bias,
    float*        __restrict__ Yout,
    float*        __restrict__ OutX,
    int mode) {
    const int lane = threadIdx.x & 31;
    const int wave = threadIdx.x >> 5;
    const int m0 = blockIdx.y * 32 + (wave >> 2) * 16;
    const int n0 = blockIdx.x * 256 + (wave & 3) * 64;
    const int r15 = lane & 15;
    const int half = lane >> 4;
    const int akb = half ? 8 : 0;      // A K-base for this lane half
    const int bkb = half ? 16 : 0;     // unused: B loads use akb, same k order as A

    const bf16_t* arow = Ab + (size_t)(m0 + r15) * CDIM;
    const bf16_t* brow = Wb + (size_t)(n0 + r15) * CDIM + akb;

    v8f acc[4];
#pragma unroll
    for (int nt = 0; nt < 4; ++nt) acc[nt] = (v8f){0.f,0.f,0.f,0.f,0.f,0.f,0.f,0.f};

#pragma unroll 2
    for (int kk = 0; kk < CDIM; kk += 32) {
        v16bf a = make_frag(*(const v8bf*)(arow + kk + akb),
                            *(const v8bf*)(arow + kk + 16 + akb));
        v16bf bfrag[4];
#pragma unroll
        for (int nt = 0; nt < 4; ++nt) {
            const bf16_t* bp = brow + (size_t)nt * 16 * CDIM + kk;
            bfrag[nt] = make_frag(*(const v8bf*)(bp), *(const v8bf*)(bp + 16));
        }
#pragma unroll
        for (int nt = 0; nt < 4; ++nt)
            acc[nt] = wmma_bf16(a, bfrag[nt], acc[nt]);
    }

    // epilogue: rows m0..m0+15 cross at most one batch boundary
    const int bidx0 = m0 / NTOK;
    const int tok0  = m0 - bidx0 * NTOK;
#pragma unroll
    for (int nt = 0; nt < 4; ++nt) {
        const int col = n0 + nt * 16 + r15;
        const float bv = bias[col];
#pragma unroll
        for (int r = 0; r < 8; ++r) {
            int dm = r + half * 8;
            int m = m0 + dm;
            if (m >= TOTM) continue;
            int bidx = bidx0, tok = tok0 + dm;
            if (tok >= NTOK) { ++bidx; tok -= NTOK; }
            float val = acc[nt][r] + bv;
            if (mode == 0) {
                Yout[((size_t)bidx * NTOK + tok) * CDIM + col] = val;
            } else {
                if (tok == 0)
                    OutX[XOUT_ELEMS + (size_t)bidx * CDIM + col] = val;
                else
                    OutX[((size_t)bidx * CDIM + col) * NSPAT + (tok - 1)] = val;
            }
        }
    }
}

// -------------------- depthwise conv3d + channel LayerNorm -----------------
// lin: (B, NTOK, C) f32. One block per (b, out token).
// transposed==0: heads[((b*12+h)*Lpad + 1 + tok)*64 + d]
// transposed==1: heads[((b*12+h)*64 + d)*Lpad + 1 + tok]   (for V)
__global__ __launch_bounds__(256) void pool_ln_kernel(
    const float* __restrict__ lin,
    const float* __restrict__ conv_w,   // (C, 27)
    const float* __restrict__ ln_w,
    const float* __restrict__ ln_b,
    bf16_t*      __restrict__ heads,
    int OH, int OW, int SH, int SW, int Lpad, int transposed) {
    __shared__ float vals[CDIM];
    __shared__ float rsum[256];
    __shared__ float rsq[256];

    const int ntok = 8 * OH * OW;
    const int b  = blockIdx.x / ntok;
    const int ot = blockIdx.x % ntok;
    const int ol = ot / (OH * OW);
    const int rem = ot % (OH * OW);
    const int oh = rem / OW, ow = rem % OW;
    const int tid = threadIdx.x;

    float psum = 0.f, psq = 0.f;
    for (int c = tid; c < CDIM; c += 256) {
        float s = 0.f;
#pragma unroll
        for (int kd = 0; kd < 3; ++kd)
#pragma unroll
            for (int kh = 0; kh < 3; ++kh)
#pragma unroll
                for (int kw = 0; kw < 3; ++kw) {
                    int il = ol + kd - 1;
                    int ih = oh * SH + kh - 1;
                    int iw = ow * SW + kw - 1;
                    if (il >= 0 && il < 8 && ih >= 0 && ih < 14 && iw >= 0 && iw < 14) {
                        int isp = (il * 14 + ih) * 14 + iw;
                        s += lin[((size_t)b * NTOK + 1 + isp) * CDIM + c] *
                             conv_w[c * 27 + kd * 9 + kh * 3 + kw];
                    }
                }
        vals[c] = s;
        psum += s;
        psq += s * s;
    }
    rsum[tid] = psum;
    rsq[tid]  = psq;
    __syncthreads();
    for (int s = 128; s > 0; s >>= 1) {
        if (tid < s) { rsum[tid] += rsum[tid + s]; rsq[tid] += rsq[tid + s]; }
        __syncthreads();
    }
    const float mu  = rsum[0] * (1.f / CDIM);
    const float var = rsq[0] * (1.f / CDIM) - mu * mu;
    const float rs  = rsqrtf(var + 1e-6f);
    __syncthreads();
    for (int c = tid; c < CDIM; c += 256) {
        float y = (vals[c] - mu) * rs * ln_w[c] + ln_b[c];
        int h = c >> 6, d = c & 63;
        size_t base = ((size_t)b * HEADS + h);
        if (transposed)
            heads[(base * HD + d) * Lpad + 1 + ot] = (bf16_t)y;
        else
            heads[(base * Lpad + 1 + ot) * HD + d] = (bf16_t)y;
    }
}

// cls token (no pooling/LN) -> slot 0 of head layout
__global__ __launch_bounds__(256) void cls_to_heads_kernel(
    const float* __restrict__ lin, bf16_t* __restrict__ heads,
    int Lpad, int transposed) {
    int i = blockIdx.x * 256 + threadIdx.x;
    if (i >= BATCH * CDIM) return;
    int b = i / CDIM, c = i % CDIM;
    float v = lin[(size_t)b * NTOK * CDIM + c];   // token 0
    int h = c >> 6, d = c & 63;
    size_t base = ((size_t)b * HEADS + h);
    if (transposed)
        heads[(base * HD + d) * Lpad] = (bf16_t)v;
    else
        heads[(base * Lpad) * HD + d] = (bf16_t)v;
}

// ------------------------------ attention ----------------------------------
// One wave per block: 16 q rows of one (b, head).
// hq (B,12,NTOKP,64), hk (B,12,LKP,64), hvT (B,12,64,LKP); pads are zero.
// Output written as bf16 into dense (MPAD x CDIM) buffer for the proj GEMM.
__global__ __launch_bounds__(32) void attn_kernel(
    const bf16_t* __restrict__ hq,
    const bf16_t* __restrict__ hk,
    const bf16_t* __restrict__ hvT,
    bf16_t*       __restrict__ obf) {
    __shared__ float sc[16][LKP];

    const int lane = threadIdx.x;
    const int qt = blockIdx.x;
    const int h  = blockIdx.y;
    const int b  = blockIdx.z;
    const int q0 = qt * 16;
    const int r15 = lane & 15;
    const int half = lane >> 4;
    const int akb = half ? 8 : 0;
    const int bkb = half ? 16 : 0;

    const bf16_t* qrow = hq  + (((size_t)b * HEADS + h) * NTOKP + q0 + r15) * HD;
    const bf16_t* kp   = hk  + ((size_t)b * HEADS + h) * LKP * HD;
    const bf16_t* vT   = hvT + ((size_t)b * HEADS + h) * HD * LKP;

    // q A-fragments for K chunks 0..31 and 32..63 (contiguous 16B loads)
    v16bf aq[2];
#pragma unroll
    for (int t = 0; t < 2; ++t)
        aq[t] = make_frag(*(const v8bf*)(qrow + t * 32 + akb),
                          *(const v8bf*)(qrow + t * 32 + 16 + akb));

    // scores = q @ k^T -> LDS (pad K-rows are zero, so no guards)
    for (int kt = 0; kt < LKP / 16; ++kt) {
        v8f acc = {0.f, 0.f, 0.f, 0.f, 0.f, 0.f, 0.f, 0.f};
        const bf16_t* krow = kp + (size_t)(kt * 16 + r15) * HD;
        v16bf bk[2];
#pragma unroll
        for (int t = 0; t < 2; ++t)
            bk[t] = make_frag(*(const v8bf*)(krow + t * 32 + akb),
                              *(const v8bf*)(krow + t * 32 + 16 + akb));
#pragma unroll
        for (int t = 0; t < 2; ++t)
            acc = wmma_bf16(aq[t], bk[t], acc);
#pragma unroll
        for (int r = 0; r < 8; ++r)
            sc[r + half * 8][kt * 16 + r15] = acc[r] * SCALE;
    }
    __syncthreads();

    // per-row softmax over the 393 valid columns; zero the pad columns
    if (lane < 16) {
        float mx = -3.0e38f;
        for (int c = 0; c < LKV; ++c) mx = fmaxf(mx, sc[lane][c]);
        float s = 0.f;
        for (int c = 0; c < LKV; ++c) {
            float e = __expf(sc[lane][c] - mx);
            sc[lane][c] = e;
            s += e;
        }
        float inv = 1.f / s;
        for (int c = 0; c < LKV; ++c) sc[lane][c] *= inv;
        for (int c = LKV; c < LKP; ++c) sc[lane][c] = 0.f;
    }
    __syncthreads();

    // o = attn @ v  (16 x 64); vT gives contiguous B-fragment loads
    v8f oc[4];
#pragma unroll
    for (int nt = 0; nt < 4; ++nt) oc[nt] = (v8f){0.f,0.f,0.f,0.f,0.f,0.f,0.f,0.f};

    for (int kk = 0; kk < LKP; kk += 32) {
        v16bf aa;
#pragma unroll
        for (int e = 0; e < 8; ++e) {
            aa[e]     = (bf16_t)sc[r15][kk + akb + e];
            aa[e + 8] = (bf16_t)sc[r15][kk + 16 + akb + e];
        }
        v16bf bv[4];
#pragma unroll
        for (int nt = 0; nt < 4; ++nt) {
            const bf16_t* vcol = vT + (size_t)(nt * 16 + r15) * LKP + kk + akb;
            bv[nt] = make_frag(*(const v8bf*)(vcol), *(const v8bf*)(vcol + 16));
        }
#pragma unroll
        for (int nt = 0; nt < 4; ++nt)
            oc[nt] = wmma_bf16(aa, bv[nt], oc[nt]);
    }

#pragma unroll
    for (int nt = 0; nt < 4; ++nt)
#pragma unroll
        for (int r = 0; r < 8; ++r) {
            int m = q0 + r + half * 8;
            if (m < NTOK)
                obf[((size_t)b * NTOK + m) * CDIM + h * HD + nt * 16 + r15] =
                    (bf16_t)oc[nt][r];
        }
}

// ---------------------------------------------------------------------------
extern "C" void kernel_launch(void* const* d_in, const int* in_sizes, int n_in,
                              void* d_out, int out_size, void* d_ws, size_t ws_size,
                              hipStream_t stream) {
    const float* x    = (const float*)d_in[0];
    const float* cls  = (const float*)d_in[1];
    const float* Wq   = (const float*)d_in[2];
    const float* bq   = (const float*)d_in[3];
    const float* Wk   = (const float*)d_in[4];
    const float* bk   = (const float*)d_in[5];
    const float* Wv   = (const float*)d_in[6];
    const float* bv   = (const float*)d_in[7];
    const float* cwq  = (const float*)d_in[8];
    const float* cwk  = (const float*)d_in[9];
    const float* cwv  = (const float*)d_in[10];
    const float* lnqw = (const float*)d_in[11];
    const float* lnqb = (const float*)d_in[12];
    const float* lnkw = (const float*)d_in[13];
    const float* lnkb = (const float*)d_in[14];
    const float* lnvw = (const float*)d_in[15];
    const float* lnvb = (const float*)d_in[16];
    const float* Wp   = (const float*)d_in[17];
    const float* bp   = (const float*)d_in[18];
    float* out = (float*)d_out;

    // ---- workspace carve-up ----
    char* ws = (char*)d_ws;
    size_t off = 0;
    auto carve = [&](size_t bytes) { char* p = ws + off; off = (off + bytes + 255) & ~(size_t)255; return p; };

    const size_t WSZ = (size_t)CDIM * CDIM;
    bf16_t* wbf_q = (bf16_t*)carve(WSZ * sizeof(bf16_t));
    bf16_t* wbf_k = (bf16_t*)carve(WSZ * sizeof(bf16_t));
    bf16_t* wbf_v = (bf16_t*)carve(WSZ * sizeof(bf16_t));
    bf16_t* wbf_p = (bf16_t*)carve(WSZ * sizeof(bf16_t));

    bf16_t* Abf = (bf16_t*)carve((size_t)MPAD * CDIM * sizeof(bf16_t)); // packed A, reused for o

    const size_t LINSZ = (size_t)TOTM * CDIM;
    float* q_lin = (float*)carve(LINSZ * sizeof(float));
    float* k_lin = (float*)carve(LINSZ * sizeof(float));
    float* v_lin = (float*)carve(LINSZ * sizeof(float));

    const int HQ_N = BATCH * HEADS * NTOKP * HD;   // 9,732,096
    const int HK_N = BATCH * HEADS * LKP   * HD;   // 2,555,904
    bf16_t* hq  = (bf16_t*)carve((size_t)HQ_N * sizeof(bf16_t));
    bf16_t* hk  = (bf16_t*)carve((size_t)HK_N * sizeof(bf16_t));
    bf16_t* hvT = (bf16_t*)carve((size_t)HK_N * sizeof(bf16_t));

    // ---- 0. weights f32 -> bf16 ----
    {
        int n = (int)WSZ, g = (n + 255) / 256;
        cvt_bf16_kernel<<<g, 256, 0, stream>>>(Wq, wbf_q, n);
        cvt_bf16_kernel<<<g, 256, 0, stream>>>(Wk, wbf_k, n);
        cvt_bf16_kernel<<<g, 256, 0, stream>>>(Wv, wbf_v, n);
        cvt_bf16_kernel<<<g, 256, 0, stream>>>(Wp, wbf_p, n);
    }

    // ---- 1. pack activations: x + cls -> dense bf16 A (pads zeroed) ----
    pack_x_kernel<<<dim3(NSPAT / 32, CDIM / 32, BATCH), dim3(32, 8), 0, stream>>>(x, Abf);
    pack_cls_pad_kernel<<<((MPAD - TOTM) * CDIM + 255) / 256, 256, 0, stream>>>(cls, Abf);

    // ---- 2. Q/K/V linears (WMMA GEMM, mode 0) ----
    dim3 ggrid(CDIM / 256, MPAD / 32, 1);          // (3, 393)
    wmma_gemm_kernel<<<ggrid, 256, 0, stream>>>(Abf, wbf_q, bq, q_lin, nullptr, 0);
    wmma_gemm_kernel<<<ggrid, 256, 0, stream>>>(Abf, wbf_k, bk, k_lin, nullptr, 0);
    wmma_gemm_kernel<<<ggrid, 256, 0, stream>>>(Abf, wbf_v, bv, v_lin, nullptr, 0);

    // ---- 3. zero padded head buffers, then pool + cls fill ----
    zero_bf16_kernel<<<(HQ_N + 255) / 256, 256, 0, stream>>>(hq, HQ_N);
    zero_bf16_kernel<<<(HK_N + 255) / 256, 256, 0, stream>>>(hk, HK_N);
    zero_bf16_kernel<<<(HK_N + 255) / 256, 256, 0, stream>>>(hvT, HK_N);

    pool_ln_kernel<<<BATCH * NSPAT, 256, 0, stream>>>(q_lin, cwq, lnqw, lnqb, hq, 14, 14, 1, 1, NTOKP, 0);
    pool_ln_kernel<<<BATCH * 392,   256, 0, stream>>>(k_lin, cwk, lnkw, lnkb, hk, 7, 7, 2, 2, LKP, 0);
    pool_ln_kernel<<<BATCH * 392,   256, 0, stream>>>(v_lin, cwv, lnvw, lnvb, hvT, 7, 7, 2, 2, LKP, 1);

    {
        int g = (BATCH * CDIM + 255) / 256;
        cls_to_heads_kernel<<<g, 256, 0, stream>>>(q_lin, hq, NTOKP, 0);
        cls_to_heads_kernel<<<g, 256, 0, stream>>>(k_lin, hk, LKP, 0);
        cls_to_heads_kernel<<<g, 256, 0, stream>>>(v_lin, hvT, LKP, 1);
    }

    // ---- 4. attention -> bf16 o into Abf (pad rows still zero) ----
    dim3 agrid(NTOKP / 16, HEADS, BATCH);          // (99, 12, 8)
    attn_kernel<<<agrid, 32, 0, stream>>>(hq, hk, hvT, Abf);

    // ---- 5. output projection (WMMA GEMM, mode 1) -> d_out ----
    wmma_gemm_kernel<<<ggrid, 256, 0, stream>>>(Abf, wbf_p, bp, nullptr, out, 1);

    (void)in_sizes; (void)n_in; (void)out_size; (void)ws_size;
}
